// CrossAttention_11716670783594
// MI455X (gfx1250) — matrix operand order repaired and checked by reading, not compiled, not run on hardware
//
#include <hip/hip_runtime.h>
#include <hip/hip_bf16.h>

// ---------------------------------------------------------------------------
// CrossAttention for MI455X (gfx1250): bf16 WMMA (v_wmma_f32_16x16x32_bf16),
// fp32 accumulate, flash-style online softmax, all operand tiles fed from
// global/L2 with 16B vector accesses.  Workspace: 32 MB.
// GEMM k-loops are unrolled x2 into a structural ping-pong (two distinct
// stage register sets, peeled epilogue) so the double buffer costs zero
// register copies and loads of one stage overlap WMMAs of the other.
// ---------------------------------------------------------------------------

#define CDIM   1024   // model dim
#define NSEQ   2048   // tokens per batch
#define NTOK   4096   // B * NSEQ
#define NHEADS 16
#define HDIM   64

typedef __attribute__((ext_vector_type(16))) __bf16 v16bf;
typedef __attribute__((ext_vector_type(8)))  float  v8f;

union FragU {
    v16bf   v;
    float4  f4[2];
    unsigned u[8];
    __bf16  e[16];
};

union Pack8 {
    __bf16 e[8];
    float4 f4;
};

// ---- WMMA wrapper ---------------------------------------------------------
__device__ __forceinline__ v8f wmma_bf16(v16bf a, v16bf b, v8f c) {
    // (neg_a, A, neg_b, B, c_mod, C, reuse_a, reuse_b)
    return __builtin_amdgcn_wmma_f32_16x16x32_bf16(false, a, false, b,
                                                   (short)0, c, false, false);
}

// ---- Fragment loaders -----------------------------------------------------
// A-operand (16x32, MxK), contiguous k order: lane r holds row r; per lane:
// chunk0 = cols [16h..16h+7], chunk1 = cols [16h+8..16h+15].
__device__ __forceinline__ v16bf load_a_bf16(const __bf16* tile, int ld, int r, int h) {
    FragU f;
    const __bf16* p = tile + r * ld;
    f.f4[0] = *reinterpret_cast<const float4*>(p + 16 * h);
    f.f4[1] = *reinterpret_cast<const float4*>(p + 16 * h + 8);
    return f.v;
}

// B-operand (32x16, KxN) loaded from a row-major [n][k] source (i.e. B^T):
// lane r holds column n=r; per lane half: K = [16h .. 16h+15], packed linearly.
__device__ __forceinline__ v16bf load_b_bf16(const __bf16* tile, int ld, int r, int h) {
    FragU f;
    const __bf16* p = tile + r * ld + 16 * h;
    f.f4[0] = *reinterpret_cast<const float4*>(p);
    f.f4[1] = *reinterpret_cast<const float4*>(p + 8);
    return f.v;
}

// Same loaders but converting fp32 -> bf16 on the fly.
__device__ __forceinline__ v16bf load_a_f32(const float* tile, int ld, int r, int h) {
    FragU f;
    const float* p = tile + r * ld;
    float4 c0 = *reinterpret_cast<const float4*>(p + 16 * h);
    float4 c1 = *reinterpret_cast<const float4*>(p + 16 * h + 4);
    float4 c2 = *reinterpret_cast<const float4*>(p + 16 * h + 8);
    float4 c3 = *reinterpret_cast<const float4*>(p + 16 * h + 12);
    f.e[0]=(__bf16)c0.x; f.e[1]=(__bf16)c0.y; f.e[2]=(__bf16)c0.z; f.e[3]=(__bf16)c0.w;
    f.e[4]=(__bf16)c1.x; f.e[5]=(__bf16)c1.y; f.e[6]=(__bf16)c1.z; f.e[7]=(__bf16)c1.w;
    f.e[8]=(__bf16)c2.x; f.e[9]=(__bf16)c2.y; f.e[10]=(__bf16)c2.z; f.e[11]=(__bf16)c2.w;
    f.e[12]=(__bf16)c3.x; f.e[13]=(__bf16)c3.y; f.e[14]=(__bf16)c3.z; f.e[15]=(__bf16)c3.w;
    return f.v;
}

__device__ __forceinline__ v16bf load_b_f32(const float* tile, int ld, int r, int h) {
    FragU f;
    const float* p = tile + r * ld + 16 * h;
    float4 c0 = *reinterpret_cast<const float4*>(p);
    float4 c1 = *reinterpret_cast<const float4*>(p + 4);
    float4 c2 = *reinterpret_cast<const float4*>(p + 8);
    float4 c3 = *reinterpret_cast<const float4*>(p + 12);
    f.e[0]=(__bf16)c0.x; f.e[1]=(__bf16)c0.y; f.e[2]=(__bf16)c0.z; f.e[3]=(__bf16)c0.w;
    f.e[4]=(__bf16)c1.x; f.e[5]=(__bf16)c1.y; f.e[6]=(__bf16)c1.z; f.e[7]=(__bf16)c1.w;
    f.e[8]=(__bf16)c2.x; f.e[9]=(__bf16)c2.y; f.e[10]=(__bf16)c2.z; f.e[11]=(__bf16)c2.w;
    f.e[12]=(__bf16)c3.x; f.e[13]=(__bf16)c3.y; f.e[14]=(__bf16)c3.z; f.e[15]=(__bf16)c3.w;
    return f.v;
}

// ---- GEMM stage = A fragment + four B fragments ---------------------------
struct Stage {
    v16bf a;
    v16bf b[4];
};

// A from fp32 Wsrc row-major, B from fp32 Xsrc row-major (4 tiles of 16 rows).
__device__ __forceinline__ void stage_load_ff(Stage& s, const float* Wsrc,
                                              const float* Xsrc, int kk,
                                              int r, int h) {
    s.a = load_a_f32(Wsrc + kk, CDIM, r, h);
    #pragma unroll
    for (int t = 0; t < 4; ++t)
        s.b[t] = load_b_f32(Xsrc + (size_t)(16 * t) * CDIM + kk, CDIM, r, h);
}

// A from fp32, B from bf16 source.
__device__ __forceinline__ void stage_load_fb(Stage& s, const float* Wsrc,
                                              const __bf16* Xsrc, int kk,
                                              int r, int h) {
    s.a = load_a_f32(Wsrc + kk, CDIM, r, h);
    #pragma unroll
    for (int t = 0; t < 4; ++t)
        s.b[t] = load_b_bf16(Xsrc + (size_t)(16 * t) * CDIM + kk, CDIM, r, h);
}

__device__ __forceinline__ void stage_wmma(v8f acc[4], const Stage& s) {
    #pragma unroll
    for (int t = 0; t < 4; ++t)
        acc[t] = wmma_bf16(s.a, s.b[t], acc[t]);
}

// ---------------------------------------------------------------------------
// Kernel 1: Q/K projection, computed transposed: out^T = W * X^T.
//   m = c_out (1024, tiles of 16), n = token (4096, 64 per wave).
// ---------------------------------------------------------------------------
__global__ __launch_bounds__(256) void proj_qk_kernel(
    const float* __restrict__ X, const float* __restrict__ W,
    const float* __restrict__ bias, __bf16* __restrict__ out) {
    const int wid  = (blockIdx.x * blockDim.x + threadIdx.x) >> 5;
    const int lane = threadIdx.x & 31;
    const int r = lane & 15, h = lane >> 4;
    const int mBase = (wid & 63) << 4;      // c_out tile
    const int nBase = (wid >> 6) << 6;      // token tile (64 wide)
    const float* Wp = W + (size_t)mBase * CDIM;
    const float* Xp = X + (size_t)nBase * CDIM;

    v8f acc[4];
    #pragma unroll
    for (int t = 0; t < 4; ++t) acc[t] = v8f{};

    Stage s0, s1;
    stage_load_ff(s0, Wp, Xp, 0, r, h);
    for (int kk = 0; kk + 64 < CDIM; kk += 64) {
        stage_load_ff(s1, Wp, Xp, kk + 32, r, h);
        stage_wmma(acc, s0);
        stage_load_ff(s0, Wp, Xp, kk + 64, r, h);
        stage_wmma(acc, s1);
    }
    stage_load_ff(s1, Wp, Xp, CDIM - 32, r, h);
    stage_wmma(acc, s0);
    stage_wmma(acc, s1);

    const int c0 = mBase + 8 * h;           // first of 8 consecutive channels
    const int head = c0 >> 6, d0 = c0 & 63;
    #pragma unroll
    for (int t = 0; t < 4; ++t) {
        const int tok = nBase + 16 * t + r;
        const int b = tok >> 11, nq = tok & (NSEQ - 1);
        Pack8 s;
        #pragma unroll
        for (int i = 0; i < 8; ++i)
            s.e[i] = (__bf16)(acc[t][i] + bias[c0 + i]);
        *reinterpret_cast<float4*>(
            out + (((size_t)(b * NHEADS + head) * NSEQ + nq) * HDIM + d0)) = s.f4;
    }
}

// ---------------------------------------------------------------------------
// Kernel 2: V projection, normal orientation: out = X * W^T, stored transposed
// into V^T[B,H,D,N] (lane = channel, vgprs = 8 consecutive tokens -> b128).
// ---------------------------------------------------------------------------
__global__ __launch_bounds__(256) void proj_v_kernel(
    const float* __restrict__ X, const float* __restrict__ W,
    const float* __restrict__ bias, __bf16* __restrict__ vT) {
    const int wid  = (blockIdx.x * blockDim.x + threadIdx.x) >> 5;
    const int lane = threadIdx.x & 31;
    const int r = lane & 15, h = lane >> 4;
    const int mBase = (wid & 255) << 4;     // token tile
    const int nBase = (wid >> 8) << 6;      // c_out tile (64 wide)
    const float* Xp = X + (size_t)mBase * CDIM;   // A source
    const float* Wp = W + (size_t)nBase * CDIM;   // B source (4 tiles)

    v8f acc[4];
    #pragma unroll
    for (int t = 0; t < 4; ++t) acc[t] = v8f{};

    Stage s0, s1;
    stage_load_ff(s0, Xp, Wp, 0, r, h);
    for (int kk = 0; kk + 64 < CDIM; kk += 64) {
        stage_load_ff(s1, Xp, Wp, kk + 32, r, h);
        stage_wmma(acc, s0);
        stage_load_ff(s0, Xp, Wp, kk + 64, r, h);
        stage_wmma(acc, s1);
    }
    stage_load_ff(s1, Xp, Wp, CDIM - 32, r, h);
    stage_wmma(acc, s0);
    stage_wmma(acc, s1);

    const int tok0 = mBase + 8 * h;         // first of 8 consecutive tokens
    const int b = tok0 >> 11, n0 = tok0 & (NSEQ - 1);
    #pragma unroll
    for (int t = 0; t < 4; ++t) {
        const int c = nBase + 16 * t + r;
        const int head = c >> 6, d = c & 63;
        const float bb = bias[c];
        Pack8 s;
        #pragma unroll
        for (int i = 0; i < 8; ++i)
            s.e[i] = (__bf16)(acc[t][i] + bb);
        *reinterpret_cast<float4*>(
            vT + (((size_t)(b * NHEADS + head) * HDIM + d) * NSEQ + n0)) = s.f4;
    }
}

// ---------------------------------------------------------------------------
// Kernel 3: flash attention per wave: one (b, head, 16-row q tile).
//   S^T = K * Q^T  (A = K tile, B = Q tile)  -> lane holds a fixed q column.
//   Online softmax in log2 domain (scale folded in), then O^T += V^T * P^T.
//   All 8 fragment loads for a kv step issue at loop top; V-fragment latency
//   hides behind the S WMMAs + softmax VALU (which co-execute with XDL).
// ---------------------------------------------------------------------------
__global__ __launch_bounds__(256) void attn_kernel(
    const __bf16* __restrict__ qB, const __bf16* __restrict__ kB,
    const __bf16* __restrict__ vT, __bf16* __restrict__ obuf) {
    const int wid  = (blockIdx.x * blockDim.x + threadIdx.x) >> 5;
    const int lane = threadIdx.x & 31;
    const int r = lane & 15, h = lane >> 4;
    const int qt   = wid & 127;
    const int head = (wid >> 7) & (NHEADS - 1);
    const int b    = wid >> 11;

    const __bf16* qP = qB + ((size_t)(b * NHEADS + head) * NSEQ + qt * 16) * HDIM;
    const __bf16* kP = kB + (size_t)(b * NHEADS + head) * NSEQ * HDIM;
    const __bf16* vP = vT + (size_t)(b * NHEADS + head) * HDIM * NSEQ;

    // Q as B-operand (16 q x 64 d), loaded once.
    const v16bf bq0 = load_b_bf16(qP, HDIM, r, h);
    const v16bf bq1 = load_b_bf16(qP + 32, HDIM, r, h);

    v8f oc[4];
    #pragma unroll
    for (int t = 0; t < 4; ++t) oc[t] = v8f{};
    float run_m = -3.0e38f, run_l = 0.0f;
    const float SC = 0.125f * 1.44269504088896f;   // HEAD_DIM^-0.5 * log2(e)

    for (int kv0 = 0; kv0 < NSEQ; kv0 += 32) {
        // ---- issue all loads for this step up front --------------------
        v16bf aK00 = load_a_bf16(kP + (size_t)kv0 * HDIM, HDIM, r, h);
        v16bf aK01 = load_a_bf16(kP + (size_t)kv0 * HDIM + 32, HDIM, r, h);
        v16bf aK10 = load_a_bf16(kP + (size_t)(kv0 + 16) * HDIM, HDIM, r, h);
        v16bf aK11 = load_a_bf16(kP + (size_t)(kv0 + 16) * HDIM + 32, HDIM, r, h);
        v16bf aV[4];
        #pragma unroll
        for (int t = 0; t < 4; ++t)
            aV[t] = load_a_bf16(vP + (size_t)(16 * t) * NSEQ + kv0, NSEQ, r, h);
        if (kv0 + 32 < NSEQ) {   // prefetch next K tile (32 rows x 128B) + V cols
            __builtin_prefetch((const void*)(kP + (size_t)(kv0 + 32) * HDIM + lane * HDIM), 0, 1);
            __builtin_prefetch((const void*)(vP + (size_t)lane * NSEQ + kv0 + 32), 0, 1);
            __builtin_prefetch((const void*)(vP + (size_t)(lane + 32) * NSEQ + kv0 + 32), 0, 1);
        }

        // ---- S^T tiles: tile0 = kv [kv0..+15], tile1 = kv [kv0+16..+31] ----
        v8f s0 = v8f{}, s1 = v8f{};
        s0 = wmma_bf16(aK00, bq0, s0);
        s0 = wmma_bf16(aK01, bq1, s0);
        s1 = wmma_bf16(aK10, bq0, s1);
        s1 = wmma_bf16(aK11, bq1, s1);

        // logits (log2 domain), lane's q is fixed: per-lane stats + one shfl.
        float l0[8], l1[8];
        float mloc = -3.0e38f;
        #pragma unroll
        for (int i = 0; i < 8; ++i) {
            l0[i] = s0[i] * SC; l1[i] = s1[i] * SC;
            mloc = fmaxf(mloc, fmaxf(l0[i], l1[i]));
        }
        mloc = fmaxf(mloc, __shfl_xor(mloc, 16));
        const float nm   = fmaxf(run_m, mloc);
        const float corr = exp2f(run_m - nm);
        float p0[8], p1[8], psum = 0.0f;
        #pragma unroll
        for (int i = 0; i < 8; ++i) {
            p0[i] = exp2f(l0[i] - nm); p1[i] = exp2f(l1[i] - nm);
            psum += p0[i] + p1[i];
        }
        psum += __shfl_xor(psum, 16);
        run_l = run_l * corr + psum;
        run_m = nm;
        #pragma unroll
        for (int t = 0; t < 4; ++t)
            #pragma unroll
            for (int i = 0; i < 8; ++i) oc[t][i] *= corr;

        // Pack P^T (S^T D-layout: vgpr i <-> kv = 8h+i) into B-operand layout
        // (per lane half: kv = 16h..16h+15). Half-wave exchange of the 8 kv
        // not resident in this half, done on packed bf16 dwords.
        unsigned u0[4], u1[4];
        #pragma unroll
        for (int i = 0; i < 4; ++i) {
            Pack8 a;
            a.e[0] = (__bf16)p0[2 * i]; a.e[1] = (__bf16)p0[2 * i + 1];
            a.e[2] = (__bf16)p1[2 * i]; a.e[3] = (__bf16)p1[2 * i + 1];
            unsigned* au = reinterpret_cast<unsigned*>(a.e);
            u0[i] = au[0]; u1[i] = au[1];
        }
        FragU bp;
        #pragma unroll
        for (int i = 0; i < 4; ++i) {
            const unsigned x0 = (unsigned)__shfl_xor((int)u0[i], 16);
            const unsigned x1 = (unsigned)__shfl_xor((int)u1[i], 16);
            bp.u[i]     = h ? x1 : u0[i];   // K = 16h + 2i,2i+1
            bp.u[4 + i] = h ? u1[i] : x0;   // K = 16h + 8 + 2i,2i+1
        }

        // ---- O^T += V^T * P^T  (A = V^T tiles of 16 d-rows x 32 kv) ----
        #pragma unroll
        for (int t = 0; t < 4; ++t)
            oc[t] = wmma_bf16(aV[t], bp.v, oc[t]);
    }

    // Normalize and store O (pre-Wo) row-major bf16 [B, Nq, C].
    const float inv = 1.0f / run_l;
    const int qglob = b * NSEQ + qt * 16 + r;   // lane's token
    #pragma unroll
    for (int t = 0; t < 4; ++t) {
        Pack8 s;
        #pragma unroll
        for (int i = 0; i < 8; ++i)
            s.e[i] = (__bf16)(oc[t][i] * inv);
        const int c0 = head * HDIM + 16 * t + 8 * h;   // 8 consecutive channels
        *reinterpret_cast<float4*>(obuf + (size_t)qglob * CDIM + c0) = s.f4;
    }
}

// ---------------------------------------------------------------------------
// Kernel 4: output projection (transposed GEMM), fp32 + bias -> d_out[B,Nq,C].
// Ping-pong pipelined like kernel 1; B operand comes from bf16 O buffer.
// ---------------------------------------------------------------------------
__global__ __launch_bounds__(256) void proj_out_kernel(
    const __bf16* __restrict__ O, const float* __restrict__ Wo,
    const float* __restrict__ bias, float* __restrict__ out) {
    const int wid  = (blockIdx.x * blockDim.x + threadIdx.x) >> 5;
    const int lane = threadIdx.x & 31;
    const int r = lane & 15, h = lane >> 4;
    const int mBase = (wid & 63) << 4;      // c_out tile
    const int nBase = (wid >> 6) << 6;      // token tile
    const float*  Wp = Wo + (size_t)mBase * CDIM;
    const __bf16* Op = O  + (size_t)nBase * CDIM;

    v8f acc[4];
    #pragma unroll
    for (int t = 0; t < 4; ++t) acc[t] = v8f{};

    Stage s0, s1;
    stage_load_fb(s0, Wp, Op, 0, r, h);
    for (int kk = 0; kk + 64 < CDIM; kk += 64) {
        stage_load_fb(s1, Wp, Op, kk + 32, r, h);
        stage_wmma(acc, s0);
        stage_load_fb(s0, Wp, Op, kk + 64, r, h);
        stage_wmma(acc, s1);
    }
    stage_load_fb(s1, Wp, Op, CDIM - 32, r, h);
    stage_wmma(acc, s0);
    stage_wmma(acc, s1);

    const int c0 = mBase + 8 * h;
    #pragma unroll
    for (int t = 0; t < 4; ++t) {
        const int tok = nBase + 16 * t + r;
        float v[8];
        #pragma unroll
        for (int i = 0; i < 8; ++i) v[i] = acc[t][i] + bias[c0 + i];
        float* dst = out + (size_t)tok * CDIM + c0;
        *reinterpret_cast<float4*>(dst)     = make_float4(v[0], v[1], v[2], v[3]);
        *reinterpret_cast<float4*>(dst + 4) = make_float4(v[4], v[5], v[6], v[7]);
    }
}

// ---------------------------------------------------------------------------
extern "C" void kernel_launch(void* const* d_in, const int* in_sizes, int n_in,
                              void* d_out, int out_size, void* d_ws, size_t ws_size,
                              hipStream_t stream) {
    const float* query = (const float*)d_in[0];
    const float* key_  = (const float*)d_in[1];
    const float* value = (const float*)d_in[2];
    const float* Wq = (const float*)d_in[3];
    const float* bq = (const float*)d_in[4];
    const float* Wk = (const float*)d_in[5];
    const float* bk = (const float*)d_in[6];
    const float* Wv = (const float*)d_in[7];
    const float* bv = (const float*)d_in[8];
    const float* Wo = (const float*)d_in[9];
    const float* bo = (const float*)d_in[10];

    // Workspace layout (32 MB total, fully rewritten every call):
    char* w = (char*)d_ws;
    __bf16* qB   = (__bf16*)(w);                              // [B,H,N,D]  8 MB
    __bf16* kB   = (__bf16*)(w + (size_t)8  * 1024 * 1024);   // [B,H,N,D]  8 MB
    __bf16* vT   = (__bf16*)(w + (size_t)16 * 1024 * 1024);   // [B,H,D,N]  8 MB
    __bf16* obuf = (__bf16*)(w + (size_t)24 * 1024 * 1024);   // [B,Nq,C]   8 MB

    const dim3 blk(256);
    const dim3 grd(512);   // 4096 waves each

    proj_qk_kernel <<<grd, blk, 0, stream>>>(query, Wq, bq, qB);
    proj_qk_kernel <<<grd, blk, 0, stream>>>(key_,  Wk, bk, kB);
    proj_v_kernel  <<<grd, blk, 0, stream>>>(value, Wv, bv, vT);
    attn_kernel    <<<grd, blk, 0, stream>>>(qB, kB, vT, obuf);
    proj_out_kernel<<<grd, blk, 0, stream>>>(obuf, Wo, bo, (float*)d_out);
}
